// RelationalGAT_20177756357015
// MI455X (gfx1250) — matrix operand order repaired and checked by reading, not compiled, not run on hardware
//
#include <hip/hip_runtime.h>
#include <hip/hip_bf16.h>

typedef __attribute__((ext_vector_type(16))) _Float16 v16h;
typedef __attribute__((ext_vector_type(8)))  _Float16 v8h;
typedef __attribute__((ext_vector_type(8)))  float    v8f;

#define NND   20000
#define D_IN  256
#define HID   128
#define HEADS 4
#define D_OUT 128
#define NEG_SLOPE 0.2f

// ---------------- small utility kernels ----------------

__global__ void fill_f32(float* __restrict__ p, float v, long long n) {
  long long i = (long long)blockIdx.x * blockDim.x + threadIdx.x;
  if (i < n) p[i] = v;
}

__global__ void copy_f32(float* __restrict__ dst, const float* __restrict__ src, long long n) {
  long long i = (long long)blockIdx.x * blockDim.x + threadIdx.x;
  if (i < n) dst[i] = src[i];
}

__global__ void cvt_f32_to_f16(const float* __restrict__ src, _Float16* __restrict__ dst,
                               long long n) {
  long long i = (long long)blockIdx.x * blockDim.x + threadIdx.x;
  if (i < n) dst[i] = (_Float16)src[i];
}

// Wt[n][k] = (f16) W[k][n]   (W is [K x Nc] row-major)
__global__ void transpose_cvt(const float* __restrict__ W, _Float16* __restrict__ Wt,
                              int K, int Nc) {
  int idx = blockIdx.x * blockDim.x + threadIdx.x;
  if (idx >= K * Nc) return;
  int n = idx / K, k = idx - n * K;
  Wt[(size_t)n * K + k] = (_Float16)W[(size_t)k * Nc + n];
}

// x_mod[src[e], :] += rel_emb[edge_type[e], :]   (one block per edge, 256 ch)
__global__ void scatter_rel(const int* __restrict__ ei, const int* __restrict__ et,
                            const float* __restrict__ rel, float* __restrict__ xm,
                            int E) {
  int e = blockIdx.x;
  int s = ei[e];
  int t = et[e];
  int c = threadIdx.x;
  atomicAdd(&xm[(size_t)s * D_IN + c], rel[(size_t)t * D_IN + c]);
}

// ---------------- WMMA GEMM: C[M x Nc] = Ah[M x K] * Bt[Nc x K]^T ----------------
// Ah: f16 row-major; Bt: f16, row n of Bt = column n of B (pre-transposed).
// One wave computes a 16x64 tile: 4 accumulators, 4 v_wmma per 32-K step,
// all fragment loads are contiguous 16B (global_load_b128).
__global__ void gemm_wmma_f16v(const _Float16* __restrict__ Ah,
                               const _Float16* __restrict__ Bt,
                               float* __restrict__ C, int M, int Nc, int K) {
  int wave = (int)((blockIdx.x * blockDim.x + threadIdx.x) >> 5);
  int lane = threadIdx.x & 31;
  int ngrp = Nc >> 6;                     // 64-column groups
  int gn = wave % ngrp;
  int tm = wave / ngrp;
  if ((tm << 4) >= M) return;             // wave-uniform exit: EXEC all-1 for WMMA
  int m0 = tm << 4, n0 = gn << 6;
  int half = lane >> 4, r = lane & 15;

  v8f acc0 = {}, acc1 = {}, acc2 = {}, acc3 = {};
  const _Float16* Ap  = Ah + (size_t)(m0 + r) * K;
  const _Float16* Bp0 = Bt + (size_t)(n0 +  0 + r) * K;
  const _Float16* Bp1 = Bt + (size_t)(n0 + 16 + r) * K;
  const _Float16* Bp2 = Bt + (size_t)(n0 + 32 + r) * K;
  const _Float16* Bp3 = Bt + (size_t)(n0 + 48 + r) * K;
  int ha = half ? 16 : 0;      // A k-offset per lane-half
  int hb = half ? 16 : 0;      // B k-offset per lane-half

  for (int kk = 0; kk < K; kk += 32) {
    // A 16x32 fragment: lane holds row m0+r, k = ha..ha+15 (contiguous)
    v8h alo = *(const v8h*)(Ap + kk + ha);
    v8h ahi = *(const v8h*)(Ap + kk + 8 + ha);
    v16h a;
    #pragma unroll
    for (int i = 0; i < 8; ++i) { a[i] = alo[i]; a[8 + i] = ahi[i]; }

    // B 32x16 fragments: lane holds col n+r, k = hb..hb+15 (contiguous in Bt row)
    int kb = kk + hb;
    v16h b0, b1, b2, b3;
    {
      v8h t0 = *(const v8h*)(Bp0 + kb), t1 = *(const v8h*)(Bp0 + kb + 8);
      #pragma unroll
      for (int i = 0; i < 8; ++i) { b0[i] = t0[i]; b0[8 + i] = t1[i]; }
    }
    {
      v8h t0 = *(const v8h*)(Bp1 + kb), t1 = *(const v8h*)(Bp1 + kb + 8);
      #pragma unroll
      for (int i = 0; i < 8; ++i) { b1[i] = t0[i]; b1[8 + i] = t1[i]; }
    }
    {
      v8h t0 = *(const v8h*)(Bp2 + kb), t1 = *(const v8h*)(Bp2 + kb + 8);
      #pragma unroll
      for (int i = 0; i < 8; ++i) { b2[i] = t0[i]; b2[8 + i] = t1[i]; }
    }
    {
      v8h t0 = *(const v8h*)(Bp3 + kb), t1 = *(const v8h*)(Bp3 + kb + 8);
      #pragma unroll
      for (int i = 0; i < 8; ++i) { b3[i] = t0[i]; b3[8 + i] = t1[i]; }
    }

    acc0 = __builtin_amdgcn_wmma_f32_16x16x32_f16(false, a, false, b0, (short)0, acc0, false, false);
    acc1 = __builtin_amdgcn_wmma_f32_16x16x32_f16(false, a, false, b1, (short)0, acc1, false, false);
    acc2 = __builtin_amdgcn_wmma_f32_16x16x32_f16(false, a, false, b2, (short)0, acc2, false, false);
    acc3 = __builtin_amdgcn_wmma_f32_16x16x32_f16(false, a, false, b3, (short)0, acc3, false, false);
  }

  // C f32 layout: VGPR i -> row m0 + i + half*8, col = tile base + r
  #pragma unroll
  for (int i = 0; i < 8; ++i) {
    size_t row = (size_t)(m0 + i + half * 8) * Nc;
    C[row + n0 +  0 + r] = acc0[i];
    C[row + n0 + 16 + r] = acc1[i];
    C[row + n0 + 32 + r] = acc2[i];
    C[row + n0 + 48 + r] = acc3[i];
  }
}

// ---------------- attention projections: out[n,h] = dot(h[n,h,:], att[h,:]) ----------------
__global__ void attn_alpha(const float* __restrict__ h, const float* __restrict__ att,
                           float* __restrict__ out, int Nn, int heads, int ch) {
  int warp = (int)((blockIdx.x * blockDim.x + threadIdx.x) >> 5);
  int lane = threadIdx.x & 31;
  if (warp >= Nn * heads) return;
  int n = warp / heads, hd = warp % heads;
  const float* hp = h + (size_t)n * heads * ch + (size_t)hd * ch;
  const float* ap = att + (size_t)hd * ch;
  float s = 0.f;
  for (int c = lane; c < ch; c += 32) s += hp[c] * ap[c];
  #pragma unroll
  for (int off = 16; off > 0; off >>= 1) s += __shfl_xor(s, off, 32);
  if (lane == 0) out[warp] = s;
}

// ---------------- segment softmax over dst ----------------

__device__ inline void atomicMaxF(float* addr, float val) {
  if (val >= 0.f) atomicMax((int*)addr, __float_as_int(val));
  else            atomicMin((unsigned int*)addr, __float_as_uint(val));
}

__device__ inline void edge_sd(const int* ei, int E, int e, int& s, int& d) {
  if (e < E) { s = ei[e]; d = ei[E + e]; }
  else       { s = d = e - E; }   // self loops appended
}

__global__ void edge_e_max(const int* __restrict__ ei, int E, int heads,
                           const float* __restrict__ as, const float* __restrict__ ad,
                           float* __restrict__ ebuf, float* __restrict__ m) {
  int idx = blockIdx.x * blockDim.x + threadIdx.x;
  int E2 = E + NND;
  if (idx >= E2 * heads) return;
  int e = idx / heads, hd = idx - e * heads;
  int s, d; edge_sd(ei, E, e, s, d);
  float v = as[s * heads + hd] + ad[d * heads + hd];
  v = (v >= 0.f) ? v : NEG_SLOPE * v;
  ebuf[idx] = v;
  atomicMaxF(&m[d * heads + hd], v);
}

__global__ void edge_p_sum(const int* __restrict__ ei, int E, int heads,
                           const float* __restrict__ m, float* __restrict__ ebuf,
                           float* __restrict__ den) {
  int idx = blockIdx.x * blockDim.x + threadIdx.x;
  int E2 = E + NND;
  if (idx >= E2 * heads) return;
  int e = idx / heads, hd = idx - e * heads;
  int s, d; edge_sd(ei, E, e, s, d);
  float p = __expf(ebuf[idx] - m[d * heads + hd]);
  ebuf[idx] = p;
  atomicAdd(&den[d * heads + hd], p);
}

// out[dst, :] += h[src, :] * alpha[head]   (one block per edge)
__global__ void edge_aggregate(const int* __restrict__ ei, int E, int heads, int ch,
                               const float* __restrict__ h, const float* __restrict__ ebuf,
                               const float* __restrict__ den, float* __restrict__ out) {
  int e = blockIdx.x;
  int s, d; edge_sd(ei, E, e, s, d);
  __shared__ float sal[8];
  if ((int)threadIdx.x < heads)
    sal[threadIdx.x] = ebuf[(size_t)e * heads + threadIdx.x] /
                       (den[(size_t)d * heads + threadIdx.x] + 1e-16f);
  __syncthreads();
  int HC = heads * ch;
  for (int c = threadIdx.x; c < HC; c += blockDim.x)
    atomicAdd(&out[(size_t)d * HC + c], h[(size_t)s * HC + c] * sal[c / ch]);
}

// ---------------- bias / activation ----------------

__global__ void bias_elu(float* __restrict__ x, const float* __restrict__ b,
                         long long total, int HC) {
  long long i = (long long)blockIdx.x * blockDim.x + threadIdx.x;
  if (i >= total) return;
  int c = (int)(i % HC);
  float v = x[i] + b[c];
  x[i] = (v > 0.f) ? v : (__expf(v) - 1.f);
}

__global__ void bias_add(float* __restrict__ x, const float* __restrict__ b,
                         long long total, int HC) {
  long long i = (long long)blockIdx.x * blockDim.x + threadIdx.x;
  if (i >= total) return;
  x[i] += b[(int)(i % HC)];
}

// ---------------- orchestration ----------------

static inline int blks(long long n, int t) { return (int)((n + t - 1) / t); }

extern "C" void kernel_launch(void* const* d_in, const int* in_sizes, int n_in,
                              void* d_out, int out_size, void* d_ws, size_t ws_size,
                              hipStream_t stream) {
  const float* x        = (const float*)d_in[0];
  const int*   ei       = (const int*)  d_in[1];   // [2,E] flat: src row then dst row
  const int*   et       = (const int*)  d_in[2];
  const float* rel_emb  = (const float*)d_in[3];
  const float* W1       = (const float*)d_in[4];
  const float* att_src1 = (const float*)d_in[5];
  const float* att_dst1 = (const float*)d_in[6];
  const float* b1       = (const float*)d_in[7];
  const float* W2       = (const float*)d_in[8];
  const float* att_src2 = (const float*)d_in[9];
  const float* att_dst2 = (const float*)d_in[10];
  const float* b2       = (const float*)d_in[11];
  float* out = (float*)d_out;

  const int E  = in_sizes[1] / 2;
  const int E2 = E + NND;
  const int HC1 = HEADS * HID;     // 512

  // workspace layout (in float units; f16 buffers cast from float space)
  float* ws = (float*)d_ws;
  size_t o = 0;
  float* x_mod = ws + o; o += (size_t)NND * D_IN;        // 5.12M   (reused as h2)
  float* h1    = ws + o; o += (size_t)NND * HC1;         // 10.24M
  float* agg1  = ws + o; o += (size_t)NND * HC1;         // 10.24M
  float* ebuf  = ws + o; o += (size_t)E2 * HEADS;        // 2.64M
  _Float16* xh  = (_Float16*)(ws + o); o += (size_t)NND * D_IN / 2;   // f16 A, layer1
  _Float16* w1t = (_Float16*)(ws + o); o += (size_t)HC1 * D_IN / 2;   // f16 W1^T
  _Float16* a2h = (_Float16*)(ws + o); o += (size_t)NND * HC1 / 2;    // f16 A, layer2
  _Float16* w2t = (_Float16*)(ws + o); o += (size_t)D_OUT * HC1 / 2;  // f16 W2^T
  float* as1   = ws + o; o += (size_t)NND * HEADS;
  float* ad1   = ws + o; o += (size_t)NND * HEADS;
  float* m1    = ws + o; o += (size_t)NND * HEADS;
  float* dn1   = ws + o; o += (size_t)NND * HEADS;
  float* as2   = ws + o; o += (size_t)NND;
  float* ad2   = ws + o; o += (size_t)NND;
  float* m2    = ws + o; o += (size_t)NND;
  float* dn2   = ws + o; o += (size_t)NND;
  float* h2    = x_mod;                                   // reuse after conversion

  // 1) x_mod = x ; x_mod[src] += rel_emb[edge_type] ; convert to f16
  copy_f32<<<blks((long long)NND * D_IN, 256), 256, 0, stream>>>(x_mod, x, (long long)NND * D_IN);
  scatter_rel<<<E, D_IN, 0, stream>>>(ei, et, rel_emb, x_mod, E);
  cvt_f32_to_f16<<<blks((long long)NND * D_IN, 256), 256, 0, stream>>>(x_mod, xh, (long long)NND * D_IN);
  transpose_cvt<<<blks((long long)D_IN * HC1, 256), 256, 0, stream>>>(W1, w1t, D_IN, HC1);

  // 2) h1 = x_mod @ W1   [N,512]  (WMMA, 16x64 tile per wave)
  {
    int tiles = (NND / 16) * (HC1 / 64);
    gemm_wmma_f16v<<<(tiles + 7) / 8, 256, 0, stream>>>(xh, w1t, h1, NND, HC1, D_IN);
  }

  // 3) attention projections
  attn_alpha<<<blks((long long)NND * HEADS * 32, 256), 256, 0, stream>>>(h1, att_src1, as1, NND, HEADS, HID);
  attn_alpha<<<blks((long long)NND * HEADS * 32, 256), 256, 0, stream>>>(h1, att_dst1, ad1, NND, HEADS, HID);

  // 4) segment softmax + aggregate (layer 1)
  fill_f32<<<blks((long long)NND * HEADS, 256), 256, 0, stream>>>(m1, -INFINITY, (long long)NND * HEADS);
  fill_f32<<<blks((long long)NND * HEADS, 256), 256, 0, stream>>>(dn1, 0.f, (long long)NND * HEADS);
  fill_f32<<<blks((long long)NND * HC1, 256), 256, 0, stream>>>(agg1, 0.f, (long long)NND * HC1);
  edge_e_max<<<blks((long long)E2 * HEADS, 256), 256, 0, stream>>>(ei, E, HEADS, as1, ad1, ebuf, m1);
  edge_p_sum<<<blks((long long)E2 * HEADS, 256), 256, 0, stream>>>(ei, E, HEADS, m1, ebuf, dn1);
  edge_aggregate<<<E2, 256, 0, stream>>>(ei, E, HEADS, HID, h1, ebuf, dn1, agg1);

  // 5) bias + ELU ; convert activations + W2 to f16
  bias_elu<<<blks((long long)NND * HC1, 256), 256, 0, stream>>>(agg1, b1, (long long)NND * HC1, HC1);
  cvt_f32_to_f16<<<blks((long long)NND * HC1, 256), 256, 0, stream>>>(agg1, a2h, (long long)NND * HC1);
  transpose_cvt<<<blks((long long)HC1 * D_OUT, 256), 256, 0, stream>>>(W2, w2t, HC1, D_OUT);

  // 6) h2 = agg1 @ W2   [N,128]  (WMMA)
  {
    int tiles = (NND / 16) * (D_OUT / 64);
    gemm_wmma_f16v<<<(tiles + 7) / 8, 256, 0, stream>>>(a2h, w2t, h2, NND, D_OUT, HC1);
  }

  // 7) layer-2 attention (heads=1)
  attn_alpha<<<blks((long long)NND * 32, 256), 256, 0, stream>>>(h2, att_src2, as2, NND, 1, D_OUT);
  attn_alpha<<<blks((long long)NND * 32, 256), 256, 0, stream>>>(h2, att_dst2, ad2, NND, 1, D_OUT);

  // 8) segment softmax + aggregate (layer 2) directly into d_out
  fill_f32<<<blks(NND, 256), 256, 0, stream>>>(m2, -INFINITY, NND);
  fill_f32<<<blks(NND, 256), 256, 0, stream>>>(dn2, 0.f, NND);
  fill_f32<<<blks((long long)NND * D_OUT, 256), 256, 0, stream>>>(out, 0.f, (long long)NND * D_OUT);
  edge_e_max<<<blks((long long)E2, 256), 256, 0, stream>>>(ei, E, 1, as2, ad2, ebuf, m2);
  edge_p_sum<<<blks((long long)E2, 256), 256, 0, stream>>>(ei, E, 1, m2, ebuf, dn2);
  edge_aggregate<<<E2, 256, 0, stream>>>(ei, E, 1, D_OUT, h2, ebuf, dn2, out);

  // 9) final bias
  bias_add<<<blks((long long)NND * D_OUT, 256), 256, 0, stream>>>(out, b2, (long long)NND * D_OUT, D_OUT);
}
